// AKT_46222438040202
// MI455X (gfx1250) — matrix operand order repaired and checked by reading, not compiled, not run on hardware
//
#include <hip/hip_runtime.h>
#include <hip/hip_bf16.h>
#include <math.h>

// Problem constants (from reference)
#define B_   8
#define S_   1024
#define D_   256
#define H_   8
#define DK_  32
#define NEGV (-1.0e32f)

typedef __bf16 bf16_t;
typedef __bf16 v16bf __attribute__((ext_vector_type(16)));
typedef __bf16 v8bf  __attribute__((ext_vector_type(8)));
typedef float  v8f   __attribute__((ext_vector_type(8)));
typedef unsigned int v4u __attribute__((ext_vector_type(4)));
typedef int v8i __attribute__((ext_vector_type(8)));
typedef int v4i __attribute__((ext_vector_type(4)));

__device__ __forceinline__ v8f wmma_bf16(v16bf a, v16bf b, v8f c) {
  // D(16x16,f32) = A(16x32,bf16) x B(32x16,bf16) + C
  return __builtin_amdgcn_wmma_f32_16x16x32_bf16(
      /*neg_a=*/false, a, /*neg_b=*/false, b,
      /*c_mod=*/(short)0, c, /*reuse_a=*/false, /*reuse_b=*/false);
}

// A-matrix 16x32 bf16 fragment from a row-major f32 matrix (contiguous k order used here, not the ISA table order:
// lane m=lane&15 owns row M=m; K = g*16..+15 contiguous, g=lane>>4).
__device__ __forceinline__ v16bf load_a_f32(const float* __restrict__ base,
                                            int ld, int lane) {
  const int m = lane & 15, g = lane >> 4;
  const float* p0 = base + (size_t)m * ld + g * 16;  // K = g*16 ..
  const float* p1 = p0 + 8;                          // K = g*16+8 ..
  v16bf a;
#pragma unroll
  for (int j = 0; j < 8; ++j) { a[j] = (bf16_t)p0[j]; a[8 + j] = (bf16_t)p1[j]; }
  return a;
}

// Same A fragment but from a bf16 row-major matrix: two 16-byte loads.
__device__ __forceinline__ v16bf load_a_bf16(const bf16_t* __restrict__ base,
                                             int ld, int lane) {
  const int m = lane & 15, g = lane >> 4;
  const bf16_t* p0 = base + (size_t)m * ld + g * 16;
  v8bf lo = *reinterpret_cast<const v8bf*>(p0);
  v8bf hi = *reinterpret_cast<const v8bf*>(p0 + 8);
  return __builtin_shufflevector(lo, hi, 0, 1, 2, 3, 4, 5, 6, 7,
                                 8, 9, 10, 11, 12, 13, 14, 15);
}

// B-matrix 32x16 bf16 fragment where B[k,n] = M[n*ld + k]  (op = X * M^T).
// Lane n=lane&15 column, K = (lane>>4)*16 .. +15 contiguous.
__device__ __forceinline__ v16bf load_bT_f32(const float* __restrict__ Mrow,
                                             int ld, int lane) {
  const int n = lane & 15, g = lane >> 4;
  const float* p = Mrow + (size_t)n * ld + g * 16;
  v16bf b;
#pragma unroll
  for (int j = 0; j < 16; ++j) b[j] = (bf16_t)p[j];
  return b;
}

// bf16 source, 32-byte aligned contiguous run -> single v16bf load.
__device__ __forceinline__ v16bf load_bT_bf16(const bf16_t* __restrict__ Mrow,
                                              int ld, int lane) {
  const int n = lane & 15, g = lane >> 4;
  return *reinterpret_cast<const v16bf*>(Mrow + (size_t)n * ld + g * 16);
}

// ---------------------------------------------------------------------------
// Projection: Y = X @ W^T + bias.
// mode 0: write bf16 head layout [B,H,S,DK]        (Q, K)
// mode 1: write bf16 TRANSPOSED layout [B,H,DK,S]  (V -> contiguous B-frags)
// mode 2: write f32 flat [B*S, D]                  (output projection)
// ---------------------------------------------------------------------------
__global__ void __launch_bounds__(128)
proj_kernel(const float* __restrict__ X, const float* __restrict__ W,
            const float* __restrict__ bias,
            bf16_t* __restrict__ out_head, bf16_t* __restrict__ out_vt,
            float* __restrict__ out_flat, int mode) {
  const int lane = threadIdx.x & 31;
  const int wave = threadIdx.x >> 5;
  const int tile = blockIdx.x * 4 + wave;
  const int ncol = D_ / 16;                 // 16 column tiles
  const int rt = tile / ncol;
  const int ct = tile % ncol;

  v8f acc = {};
#pragma unroll
  for (int kb = 0; kb < D_ / 32; ++kb) {
    v16bf a = load_a_f32(X + (size_t)rt * 16 * D_ + kb * 32, D_, lane);
    v16bf b = load_bT_f32(W + (size_t)ct * 16 * D_ + kb * 32, D_, lane);
    acc = wmma_bf16(a, b, acc);
  }

  const int n = lane & 15, g = lane >> 4;
  const int col = ct * 16 + n;
  const float bv = bias[col];
#pragma unroll
  for (int r = 0; r < 8; ++r) {
    const int row = rt * 16 + r + g * 8;    // global row in [0, B*S)
    const float v = acc[r] + bv;
    if (mode == 0) {
      const int bi = row / S_, s = row % S_;
      const int h = col / DK_, dk = col % DK_;
      out_head[(((size_t)bi * H_ + h) * S_ + s) * DK_ + dk] = (bf16_t)v;
    } else if (mode == 1) {
      const int bi = row / S_, s = row % S_;
      const int h = col / DK_, dk = col % DK_;
      out_vt[(((size_t)bi * H_ + h) * DK_ + dk) * S_ + s] = (bf16_t)v;
    } else {
      out_flat[(size_t)row * D_ + col] = v;
    }
  }
}

// ---------------------------------------------------------------------------
// Fused AKT attention. LDS layout (320 KB, the WGP maximum):
//   [ K panel: S x DK bf16 = 64 KB | 4 per-wave 16x1024 f32 score strips ]
// All 4 waves of a block share one (b,h), so the K panel is loaded ONCE per
// block with the Tensor Data Mover (async, tracked by TENSORcnt) and feeds
// 4 x 64 B-fragments from LDS.
// ---------------------------------------------------------------------------
__global__ void __launch_bounds__(128)
attn_kernel(const bf16_t* __restrict__ qh, const bf16_t* __restrict__ kh,
            const bf16_t* __restrict__ vt, const float* __restrict__ gammas,
            float* __restrict__ attnout) {
  extern __shared__ float smem[];
  const int lane = threadIdx.x & 31;
  const int wave = threadIdx.x >> 5;

  bf16_t* kpanel = reinterpret_cast<bf16_t*>(smem);            // 64 KB
  float* strip = smem + (S_ * DK_ / 2) + (size_t)wave * 16 * S_; // 64 KB/wave

  const int tile = blockIdx.x * 4 + wave;
  const int qtiles = S_ / 16;               // 64
  const int bh = (blockIdx.x * 4) / qtiles; // uniform across the block
  const int qt = tile % qtiles;
  const int h = bh % H_;
  const int b = bh / H_;
  const int s0 = qt * 16;

  const bf16_t* qhp = qh + (size_t)bh * S_ * DK_;
  const bf16_t* khp = kh + (size_t)bh * S_ * DK_;
  const bf16_t* vtp = vt + (size_t)bh * DK_ * S_;

  const float inv_sqrt_dk = 0.17677669529663687f;        // 1/sqrt(32)
  const float gamma = -__logf(1.0f + __expf(gammas[h])); // -softplus(g)

  const int n = lane & 15, g = lane >> 4;

  // ---- Stage K panel (64 KB bf16) into LDS via the Tensor Data Mover ------
#if __has_builtin(__builtin_amdgcn_tensor_load_to_lds) && \
    __has_builtin(__builtin_amdgcn_s_wait_tensorcnt)
  if (wave == 0) {
    const unsigned NUNITS = (S_ * DK_ * 2) / 8;          // 8192 x 8-byte units
    const unsigned lds_off = (unsigned)(uintptr_t)(void*)kpanel;
    const unsigned long long ga = (unsigned long long)(uintptr_t)khp;
    // D# group0: count=1 | lds_addr | global_addr[56:0] | type=2
    v4u g0 = {1u, lds_off, (unsigned)(ga & 0xffffffffu),
              (unsigned)((ga >> 32) & 0x01ffffffu) | (2u << 30)};
    // D# group1: data_size=3 (8B), 1-D tile: tensor_dim0=tile_dim0=8192,
    // tensor_dim1=1, tile_dim1=1, tensor_dim0_stride=8192.
    v8i g1 = {(int)(3u << 16),
              (int)((NUNITS & 0xffffu) << 16),
              (int)((NUNITS >> 16) | (1u << 16)),
              (int)((NUNITS & 0xffffu) << 16),
              1,
              (int)NUNITS,
              0, 0};
    v4i gz4 = {0, 0, 0, 0};
    v8i gz8 = {0, 0, 0, 0, 0, 0, 0, 0};
    // 6-arg toolchain form: (g0, g1, g2, g3, g4, cpol)
    __builtin_amdgcn_tensor_load_to_lds(g0, g1, gz4, gz4, gz8, 0);
    __builtin_amdgcn_s_wait_tensorcnt(0);
  }
#else
  {
    const uint4* src = reinterpret_cast<const uint4*>(khp);
    uint4* dst = reinterpret_cast<uint4*>(kpanel);
    for (int i = threadIdx.x; i < (S_ * DK_ * 2) / 16; i += 128) dst[i] = src[i];
  }
#endif
  __syncthreads();

  // ---- Phase 1: score strip = (Q K^T)/sqrt(dk), B-frags from LDS ----------
  v16bf aq = load_a_bf16(qhp + (size_t)s0 * DK_, DK_, lane);   // whole K=DK=32
  for (int kt = 0; kt < S_ / 16; ++kt) {
    v16bf bk = *reinterpret_cast<const v16bf*>(
        kpanel + (size_t)(kt * 16 + n) * DK_ + g * 16);
    v8f c = {};
    c = wmma_bf16(aq, bk, c);
#pragma unroll
    for (int r = 0; r < 8; ++r)
      strip[(size_t)(r + g * 8) * S_ + kt * 16 + n] = c[r] * inv_sqrt_dk;
  }
  __syncthreads();

  // ---- Phase 2: monotonic-attention softmax chain, one row at a time ------
  for (int r = 0; r < 16; ++r) {
    const int sq = s0 + r;                  // query position (mask: t < sq)
    float* rowp = strip + (size_t)r * S_;
    const int t0 = lane * 32;

    if (sq == 0) {                          // zero_pad: attn[:, :, 0, :] = 0
#pragma unroll
      for (int j = 0; j < 32; ++j) rowp[t0 + j] = 0.0f;
      continue;                             // uniform across the wave
    }

    float x[32], w[32];
    const float4* r4 = reinterpret_cast<const float4*>(rowp + t0);
#pragma unroll
    for (int j = 0; j < 8; ++j) {
      float4 v = r4[j];
      x[4 * j + 0] = v.x; x[4 * j + 1] = v.y;
      x[4 * j + 2] = v.z; x[4 * j + 3] = v.w;
    }

    // masked softmax -> sm
    float mx = NEGV;
#pragma unroll
    for (int j = 0; j < 32; ++j) if (t0 + j < sq) mx = fmaxf(mx, x[j]);
#pragma unroll
    for (int d = 16; d >= 1; d >>= 1) mx = fmaxf(mx, __shfl_xor(mx, d, 32));

    float lsum = 0.0f;
#pragma unroll
    for (int j = 0; j < 32; ++j) {
      const float e = (t0 + j < sq) ? __expf(x[j] - mx) : 0.0f;
      w[j] = e; lsum += e;
    }
    float tot = lsum;
#pragma unroll
    for (int d = 16; d >= 1; d >>= 1) tot += __shfl_xor(tot, d, 32);
    const float rinv = 1.0f / tot;

    // inclusive cumsum of sm: in-chunk sequential + cross-lane scan
    float run = 0.0f;
#pragma unroll
    for (int j = 0; j < 32; ++j) { run += w[j] * rinv; w[j] = run; }
    float inc = run;
#pragma unroll
    for (int d = 1; d < 32; d <<= 1) {
      const float u = __shfl_up(inc, d, 32);
      if (lane >= d) inc += u;
    }
    const float excl = inc - run;
    const float disttot = __shfl(inc, 31, 32);

    // distance-decay rescale
#pragma unroll
    for (int j = 0; j < 32; ++j) {
      const int t = t0 + j;
      const float dc = excl + w[j];                    // distcum
      const float pe = fabsf((float)(t - sq));         // |t - s|
      float d2 = (disttot - dc) * pe;
      d2 = d2 > 0.0f ? d2 : 0.0f;
      float te = __expf(sqrtf(d2) * gamma);
      te = fminf(fmaxf(te, 1e-5f), 1e5f);
      x[j] = (t < sq) ? x[j] * te : NEGV;
    }

    // second softmax
    float mx2 = NEGV;
#pragma unroll
    for (int j = 0; j < 32; ++j) mx2 = fmaxf(mx2, x[j]);
#pragma unroll
    for (int d = 16; d >= 1; d >>= 1) mx2 = fmaxf(mx2, __shfl_xor(mx2, d, 32));
    float s2 = 0.0f;
#pragma unroll
    for (int j = 0; j < 32; ++j) { const float e = __expf(x[j] - mx2); x[j] = e; s2 += e; }
#pragma unroll
    for (int d = 16; d >= 1; d >>= 1) s2 += __shfl_xor(s2, d, 32);
    const float rinv2 = 1.0f / s2;

    float4* w4 = reinterpret_cast<float4*>(rowp + t0);
#pragma unroll
    for (int j = 0; j < 8; ++j) {
      float4 v;
      v.x = x[4 * j + 0] * rinv2; v.y = x[4 * j + 1] * rinv2;
      v.z = x[4 * j + 2] * rinv2; v.w = x[4 * j + 3] * rinv2;
      w4[j] = v;
    }
  }
  __syncthreads();

  // ---- Phase 3: out = attn @ V  (A from LDS strip, B from bf16 V^T) -------
  v8f o0 = {}, o1 = {};
  for (int kb = 0; kb < S_ / 32; ++kb) {
    if (kb + 1 < S_ / 32)
      __builtin_prefetch(vtp + (kb + 1) * 32, 0, 1);
    const float* p0 = strip + (size_t)n * S_ + kb * 32 + g * 16;
    v16bf a;
#pragma unroll
    for (int j = 0; j < 8; ++j) { a[j] = (bf16_t)p0[j]; a[8 + j] = (bf16_t)p0[8 + j]; }
    v16bf b0 = load_bT_bf16(vtp + kb * 32, S_, lane);                   // d 0..15
    v16bf b1 = load_bT_bf16(vtp + (size_t)16 * S_ + kb * 32, S_, lane); // d 16..31
    o0 = wmma_bf16(a, b0, o0);
    o1 = wmma_bf16(a, b1, o1);
  }
#pragma unroll
  for (int r = 0; r < 8; ++r) {
    const int srow = s0 + r + g * 8;
    const size_t base = ((size_t)b * S_ + srow) * D_ + h * DK_;
    attnout[base + n] = o0[r];
    attnout[base + 16 + n] = o1[r];
  }
}

// ---------------------------------------------------------------------------
extern "C" void kernel_launch(void* const* d_in, const int* in_sizes, int n_in,
                              void* d_out, int out_size, void* d_ws, size_t ws_size,
                              hipStream_t stream) {
  (void)in_sizes; (void)n_in; (void)out_size; (void)ws_size;
  // setup_inputs order: q,k,v,mask,Wq,bq,Wk,bk,Wv,bv,Wo,bo,gammas,zero_pad
  const float* q  = (const float*)d_in[0];
  const float* k  = (const float*)d_in[1];
  const float* v  = (const float*)d_in[2];
  const float* Wq = (const float*)d_in[4];
  const float* bq = (const float*)d_in[5];
  const float* Wk = (const float*)d_in[6];
  const float* bk = (const float*)d_in[7];
  const float* Wv = (const float*)d_in[8];
  const float* bv = (const float*)d_in[9];
  const float* Wo = (const float*)d_in[10];
  const float* bo = (const float*)d_in[11];
  const float* gm = (const float*)d_in[12];
  float* out = (float*)d_out;

  // Workspace: qh bf16 4MB | kh bf16 4MB | vt bf16 4MB | attnout f32 8MB
  char* ws = (char*)d_ws;
  bf16_t* qh = (bf16_t*)(ws);
  bf16_t* kh = (bf16_t*)(ws + ((size_t)4 << 20));
  bf16_t* vt = (bf16_t*)(ws + ((size_t)8 << 20));
  float*  ao = (float*)(ws + ((size_t)12 << 20));

  const dim3 blk(128);
  const int proj_blocks = ((B_ * S_) / 16) * (D_ / 16) / 4;   // 2048
  proj_kernel<<<proj_blocks, blk, 0, stream>>>(q, Wq, bq, qh, nullptr, nullptr, 0);
  proj_kernel<<<proj_blocks, blk, 0, stream>>>(k, Wk, bk, kh, nullptr, nullptr, 0);
  proj_kernel<<<proj_blocks, blk, 0, stream>>>(v, Wv, bv, nullptr, vt, nullptr, 1);

  const int attn_blocks = (B_ * H_ * (S_ / 16)) / 4;          // 1024
  const size_t lds = (size_t)(S_ * DK_ * 2) +                 // K panel 64 KB
                     (size_t)4 * 16 * S_ * sizeof(float);     // strips 256 KB
  attn_kernel<<<attn_blocks, blk, lds, stream>>>(qh, kh, vt, gm, ao);

  proj_kernel<<<proj_blocks, blk, 0, stream>>>(ao, Wo, bo, nullptr, nullptr, out, 2);
}
